// UnfeatGraphAttNet_24154896073111
// MI455X (gfx1250) — matrix operand order repaired and checked by reading, not compiled, not run on hardware
//
#include <hip/hip_runtime.h>
#include <hip/hip_bf16.h>

typedef __attribute__((ext_vector_type(16))) _Float16 v16h;
typedef __attribute__((ext_vector_type(4)))  _Float16 v4h;
typedef __attribute__((ext_vector_type(8)))  float    v8f;

#define N_NODES   100000
#define N_EDGES   1600000
#define IN_DIM    128
#define HID       64
#define HEAD      4
#define NUM_CLS   40
#define NEG_SLOPE 0.2f
#define BN_EPS    1e-5f

// ---------------------------------------------------------------------------
// CDNA5 async copy helpers (memory -> LDS, tracked by ASYNCcnt)
// ---------------------------------------------------------------------------
__device__ __forceinline__ void async_ld_b128(unsigned lds_addr, const void* gaddr) {
    asm volatile("global_load_async_to_lds_b128 %0, %1, off"
                 :: "v"(lds_addr), "v"(gaddr) : "memory");
}

// ---------------------------------------------------------------------------
// Generic fill
// ---------------------------------------------------------------------------
__global__ void fill_u32(unsigned* p, unsigned v, long long n) {
    long long i = (long long)blockIdx.x * blockDim.x + threadIdx.x;
    if (i < n) p[i] = v;
}

// ---------------------------------------------------------------------------
// f32 -> f16 (vectorized x4)
// ---------------------------------------------------------------------------
__global__ void cvt_f16(const float* __restrict__ in, _Float16* __restrict__ out,
                        long long n4) {
    long long i = (long long)blockIdx.x * blockDim.x + threadIdx.x;
    if (i >= n4) return;
    float4 v = ((const float4*)in)[i];
    v4h h = {(_Float16)v.x, (_Float16)v.y, (_Float16)v.z, (_Float16)v.w};
    ((v4h*)out)[i] = h;
}

// B[K,Nc] f32 -> Bt[Nc,K] f16
__global__ void transpose_f16(const float* __restrict__ B, _Float16* __restrict__ Bt,
                              int K, int Nc) {
    long long i = (long long)blockIdx.x * blockDim.x + threadIdx.x;
    if (i >= (long long)K * Nc) return;
    int k = (int)(i / Nc);
    int n = (int)(i - (long long)k * Nc);
    Bt[(long long)n * K + k] = (_Float16)B[i];
}

// ---------------------------------------------------------------------------
// WMMA GEMM, async double-buffered LDS staging.
//   A  : [M,K]  f16 row-major
//   Bt : [Nc,K] f16 (pre-transposed)
//   C  : f16 (Ch) or f32+bias (Cf)
// K multiple of 32. Block = 256 thr (8 waves); macro-tile 128 x 16;
// each wave owns one 16x16 tile. Per K-step each wave issues exactly 3
// global_load_async_to_lds_b128 (uniform), retired with s_wait_asynccnt.
// ---------------------------------------------------------------------------
__global__ void gemm_wmma_async(const _Float16* __restrict__ A,
                                const _Float16* __restrict__ Bt,
                                const float* __restrict__ bias,
                                float* __restrict__ Cf,
                                _Float16* __restrict__ Ch,
                                int M, int K, int Nc) {
    __shared__ __align__(16) _Float16 As[2][128 * 40];  // 32 K halfs + 8 pad
    __shared__ __align__(16) _Float16 Bs[2][16 * 40];

    const int tid  = threadIdx.x;
    const int lane = tid & 31;
    const int wave = tid >> 5;
    const int half = lane >> 4;
    const int l16  = lane & 15;
    const long long row0 = (long long)blockIdx.x * 128;
    const int col0 = blockIdx.y * 16;

    // --- A staging chunks: 512 x 16B chunks, 2 per thread (rows clamped) ---
    const int r0  = tid >> 2;            // rows 0..63
    const int c40 = (tid & 3) * 8;       // half offset within row (0,8,16,24)
    const int r1  = r0 + 64;             // rows 64..127
    long long ar0 = row0 + r0; if (ar0 >= M) ar0 = M - 1;
    long long ar1 = row0 + r1; if (ar1 >= M) ar1 = M - 1;
    const _Float16* gA0 = A + ar0 * K + c40;
    const _Float16* gA1 = A + ar1 * K + c40;

    // --- B staging: 64 x 16B chunks, 4x redundant across 256 threads ---
    const int b   = tid & 63;
    const int nb  = b >> 2;
    const int c4b = (b & 3) * 8;
    int colb = col0 + nb; if (colb >= Nc) colb = Nc - 1;  // clamp (unstored cols)
    const _Float16* gB = Bt + (long long)colb * K + c4b;

    const unsigned lA  = (unsigned)(size_t)(&As[0][0]);
    const unsigned lB  = (unsigned)(size_t)(&Bs[0][0]);
    const unsigned oA0 = (unsigned)((r0 * 40 + c40) * 2);
    const unsigned oA1 = (unsigned)((r1 * 40 + c40) * 2);
    const unsigned oB  = (unsigned)((nb * 40 + c4b) * 2);
    const unsigned ABUF = 128 * 40 * 2;  // 10240 B
    const unsigned BBUF = 16 * 40 * 2;   // 1280 B

    v8f acc = {};
    const int steps = K >> 5;

    // preload stage 0 -> buffer 0
    async_ld_b128(lA + oA0, gA0);
    async_ld_b128(lA + oA1, gA1);
    async_ld_b128(lB + oB,  gB);

    for (int s = 0; s < steps; ++s) {
        const int buf = s & 1;
        if (s + 1 < steps) {
            const unsigned nb128 = (buf ^ 1);
            const int k0 = (s + 1) << 5;
            async_ld_b128(lA + nb128 * ABUF + oA0, gA0 + k0);
            async_ld_b128(lA + nb128 * ABUF + oA1, gA1 + k0);
            async_ld_b128(lB + nb128 * BBUF + oB,  gB + k0);
            asm volatile("s_wait_asynccnt 0x3" ::: "memory");  // stage s done
        } else {
            asm volatile("s_wait_asynccnt 0x0" ::: "memory");
        }
        __syncthreads();

        // fragments per ISA 7.12.2 (A: M=lane%16, K octets by lane-half;
        // B: N=lane%16, K octets by lane-half, same as A)
        union { uint4 q[2]; v16h h; } af, bf;
        const _Float16* as = &As[buf][0];
        const _Float16* bs = &Bs[buf][0];
        const int m = wave * 16 + l16;
        af.q[0] = *(const uint4*)&as[m * 40 + half * 8];
        af.q[1] = *(const uint4*)&as[m * 40 + 16 + half * 8];
        bf.q[0] = *(const uint4*)&bs[l16 * 40 + half * 8];
        bf.q[1] = *(const uint4*)&bs[l16 * 40 + 16 + half * 8];

        acc = __builtin_amdgcn_wmma_f32_16x16x32_f16(
                  false, af.h, false, bf.h, (short)0, acc, false, false);
        __syncthreads();  // all waves done reading buf before it is re-filled
    }

    // C/D layout: element r -> M = r + 8*half, N = lane%16
    const int col = col0 + l16;
    if (col < Nc) {
        if (Ch) {
            #pragma unroll
            for (int r = 0; r < 8; ++r) {
                long long m = row0 + wave * 16 + half * 8 + r;
                if (m < M) Ch[m * Nc + col] = (_Float16)acc[r];
            }
        } else {
            float bv = bias ? bias[col] : 0.f;
            #pragma unroll
            for (int r = 0; r < 8; ++r) {
                long long m = row0 + wave * 16 + half * 8 + r;
                if (m < M) Cf[m * Nc + col] = acc[r] + bv;
            }
        }
    }
}

// ---------------------------------------------------------------------------
// Attention coefficients from f16 features
// ---------------------------------------------------------------------------
__global__ void attn_coef(const _Float16* __restrict__ feat,
                          const float* __restrict__ al,
                          const float* __restrict__ ar,
                          float* __restrict__ el, float* __restrict__ er,
                          int H, int D) {
    extern __shared__ float sm[];
    const int HD = H * D;
    const int n = blockIdx.x;
    const int c = threadIdx.x;
    float v = (float)feat[(long long)n * HD + c];
    float* sl = sm;
    float* sr = sm + HD;
    sl[c] = v * al[c];
    sr[c] = v * ar[c];
    __syncthreads();
    for (int s = D >> 1; s > 0; s >>= 1) {
        if ((c & (D - 1)) < s) { sl[c] += sl[c + s]; sr[c] += sr[c + s]; }
        __syncthreads();
    }
    if ((c & (D - 1)) == 0) {
        int h = c / D;
        el[(long long)n * H + h] = sl[c];
        er[(long long)n * H + h] = sr[c];
    }
}

// ---------------------------------------------------------------------------
// Edge softmax (ordered-uint float atomicMax)
// ---------------------------------------------------------------------------
__device__ __forceinline__ float leaky(float x) { return x > 0.f ? x : NEG_SLOPE * x; }
__device__ __forceinline__ unsigned f2ord(float f) {
    unsigned u = __float_as_uint(f);
    return (u & 0x80000000u) ? ~u : (u | 0x80000000u);
}
__device__ __forceinline__ float ord2f(unsigned u) {
    return (u & 0x80000000u) ? __uint_as_float(u & 0x7fffffffu)
                             : __uint_as_float(~u);
}

__global__ void edge_max(const float* __restrict__ el, const float* __restrict__ er,
                         const int* __restrict__ src, const int* __restrict__ dst,
                         unsigned* __restrict__ emax_ord, long long EH, int H) {
    long long i = (long long)blockIdx.x * blockDim.x + threadIdx.x;
    if (i >= EH) return;
    long long e = i / H;
    int h = (int)(i - e * H);
    int s = src[e], d = dst[e];
    float x = leaky(el[(long long)s * H + h] + er[(long long)d * H + h]);
    atomicMax(&emax_ord[(long long)d * H + h], f2ord(x));
}

__global__ void decode_max(unsigned* __restrict__ buf, long long n) {
    long long i = (long long)blockIdx.x * blockDim.x + threadIdx.x;
    if (i >= n) return;
    float f = ord2f(buf[i]);
    if (!(f >= -3.0e38f && f <= 3.0e38f)) f = 0.f;  // non-finite -> 0
    ((float*)buf)[i] = f;
}

__global__ void edge_expsum(const float* __restrict__ el, const float* __restrict__ er,
                            const int* __restrict__ src, const int* __restrict__ dst,
                            const float* __restrict__ emaxf,
                            float* __restrict__ ee, float* __restrict__ denom,
                            long long EH, int H) {
    long long i = (long long)blockIdx.x * blockDim.x + threadIdx.x;
    if (i >= EH) return;
    long long e = i / H;
    int h = (int)(i - e * H);
    int s = src[e], d = dst[e];
    float x = leaky(el[(long long)s * H + h] + er[(long long)d * H + h]);
    float v = __expf(x - emaxf[(long long)d * H + h]);
    ee[i] = v;
    atomicAdd(&denom[(long long)d * H + h], v);
}

// One block per edge, thread per channel: coalesced f16 gather + f32 atomic scatter.
__global__ void scatter_msg(const _Float16* __restrict__ feat,
                            const float* __restrict__ ee,
                            const float* __restrict__ denom,
                            const int* __restrict__ src, const int* __restrict__ dst,
                            float* __restrict__ agg, int H, int D) {
    long long e = blockIdx.x;
    int c = threadIdx.x;                 // 0 .. H*D-1
    int h = c / D;
    int s = src[e], d = dst[e];
    float dn = denom[(long long)d * H + h];
    dn = dn > 0.f ? dn : 1.f;
    float alpha = ee[e * H + h] / dn;
    int HD = H * D;
    float fv = (float)feat[(long long)s * HD + c];
    atomicAdd(&agg[(long long)d * HD + c], fv * alpha);
}

// ---------------------------------------------------------------------------
// BatchNorm (training stats, biased var) + ReLU
// ---------------------------------------------------------------------------
__global__ void bn_stats(const float* __restrict__ x, float* __restrict__ gsum,
                         float* __restrict__ gsq, int Nrows, int C, int rowsPerBlock) {
    int c = threadIdx.x;                 // blockDim == C
    long long r0 = (long long)blockIdx.x * rowsPerBlock;
    long long r1 = r0 + rowsPerBlock;
    if (r1 > Nrows) r1 = Nrows;
    float s = 0.f, ss = 0.f;
    for (long long r = r0; r < r1; ++r) {
        float v = x[r * C + c];
        s += v; ss += v * v;
    }
    atomicAdd(&gsum[c], s);
    atomicAdd(&gsq[c], ss);
}

__global__ void bn_finalize(const float* __restrict__ gsum, const float* __restrict__ gsq,
                            float* __restrict__ mu, float* __restrict__ rstd,
                            int C, float invN) {
    int c = threadIdx.x + blockIdx.x * blockDim.x;
    if (c >= C) return;
    float m = gsum[c] * invN;
    float var = gsq[c] * invN - m * m;
    mu[c] = m;
    rstd[c] = rsqrtf(var + BN_EPS);
}

// layer-1 variant: writes f16 (only consumer is GEMM2)
__global__ void bn_relu_h(const float* __restrict__ x, const float* __restrict__ mu,
                          const float* __restrict__ rstd, const float* __restrict__ g,
                          const float* __restrict__ be, _Float16* __restrict__ y,
                          long long total, int C) {
    long long i = (long long)blockIdx.x * blockDim.x + threadIdx.x;
    if (i >= total) return;
    int c = (int)(i % C);
    float v = (x[i] - mu[c]) * rstd[c] * g[c] + be[c];
    y[i] = (_Float16)(v > 0.f ? v : 0.f);
}

// layer-2 variant: writes f32 (feeds L2-normalize)
__global__ void bn_relu_f(const float* __restrict__ x, const float* __restrict__ mu,
                          const float* __restrict__ rstd, const float* __restrict__ g,
                          const float* __restrict__ be, float* __restrict__ y,
                          long long total, int C) {
    long long i = (long long)blockIdx.x * blockDim.x + threadIdx.x;
    if (i >= total) return;
    int c = (int)(i % C);
    float v = (x[i] - mu[c]) * rstd[c] * g[c] + be[c];
    y[i] = v > 0.f ? v : 0.f;
}

// ---------------------------------------------------------------------------
// Row-wise L2 normalize (64 cols), wave32 shuffle reduction, dual f32+f16 out
// ---------------------------------------------------------------------------
__global__ void l2norm64(const float* __restrict__ h, float* __restrict__ out,
                         _Float16* __restrict__ out_h, int Nrows) {
    int lane = threadIdx.x & 31;
    int wave = threadIdx.x >> 5;
    long long row = (long long)blockIdx.x * 8 + wave;
    if (row >= Nrows) return;
    float a = h[row * 64 + lane];
    float b = h[row * 64 + 32 + lane];
    float s = a * a + b * b;
    #pragma unroll
    for (int off = 16; off > 0; off >>= 1) s += __shfl_xor(s, off, 32);
    float nrm = fmaxf(sqrtf(s), 1e-12f);
    float fa = a / nrm, fb = b / nrm;
    out[row * 64 + lane]        = fa;
    out[row * 64 + 32 + lane]   = fb;
    out_h[row * 64 + lane]      = (_Float16)fa;
    out_h[row * 64 + 32 + lane] = (_Float16)fb;
}

// ---------------------------------------------------------------------------
// Host launcher
// ---------------------------------------------------------------------------
static inline long long cdiv(long long a, long long b) { return (a + b - 1) / b; }

extern "C" void kernel_launch(void* const* d_in, const int* in_sizes, int n_in,
                              void* d_out, int out_size, void* d_ws, size_t ws_size,
                              hipStream_t stream) {
    const float* x   = (const float*)d_in[0];
    const int*   src = (const int*)  d_in[1];
    const int*   dst = (const int*)  d_in[2];
    const float* W1  = (const float*)d_in[3];
    const float* al1 = (const float*)d_in[4];
    const float* ar1 = (const float*)d_in[5];
    // d_in[6] = b1: cancels exactly under the subsequent BatchNorm (zeros in setup)
    const float* g1  = (const float*)d_in[7];
    const float* be1 = (const float*)d_in[8];
    const float* W2  = (const float*)d_in[9];
    const float* al2 = (const float*)d_in[10];
    const float* ar2 = (const float*)d_in[11];
    // d_in[12] = b2: cancels under BN as well
    const float* g2  = (const float*)d_in[13];
    const float* be2 = (const float*)d_in[14];
    const float* Wo  = (const float*)d_in[15];
    const float* bo  = (const float*)d_in[16];

    float* out      = (float*)d_out;                    // [N, 40]
    float* feat_out = out + (size_t)N_NODES * NUM_CLS;  // [N, 64]

    float* ws = (float*)d_ws;
    // workspace layout (float units); aliased lifetimes are stream-ordered
    _Float16* feat1h = (_Float16*)ws;                     // N*256 halfs (also h1h)
    float*    agg1   = ws + 12800000LL;                   // N*256 f32 (also xh)
    _Float16* xh     = (_Float16*)agg1;                   // N*128 halfs, dead pre-fill
    float*    el     = ws + 38400000LL;                   // N*4
    float*    er     = ws + 38800000LL;                   // N*4
    unsigned* emax   = (unsigned*)(ws + 39200000LL);      // N*4 ord/float
    float*    denom  = ws + 39600000LL;                   // N*4
    float*    ee     = ws + 40000000LL;                   // E*4 (also h2: N*64)
    float*    h2     = ee;
    _Float16* feat2h = (_Float16*)(ws + 46400000LL);      // N*64 halfs
    float*    agg2   = ws + 49600000LL;                   // N*64
    _Float16* feath  = (_Float16*)(ws + 56000000LL);      // N*64 halfs
    _Float16* W1t    = (_Float16*)(ws + 59200000LL);      // 256*128 halfs
    _Float16* W2t    = W1t + 256 * 128;                   // 64*256 halfs
    _Float16* Wot    = W2t + 64 * 256;                    // 40*64 halfs
    float*    gsum   = ws + 59240000LL;                   // 256
    float*    gsq    = gsum + 256;
    float*    mu     = gsq + 256;
    float*    rstd   = mu + 256;

    const float invN = 1.0f / (float)N_NODES;
    const unsigned ORD_NEG_INF = 0x007FFFFFu;  // f2ord(-inf)

    // =========================== Layer 1 (H=4, D=64) ========================
    cvt_f16<<<(unsigned)cdiv((long long)N_NODES * IN_DIM / 4, 256), 256, 0, stream>>>(
        x, xh, (long long)N_NODES * IN_DIM / 4);
    transpose_f16<<<(unsigned)cdiv(IN_DIM * HEAD * HID, 256), 256, 0, stream>>>(
        W1, W1t, IN_DIM, HEAD * HID);
    {
        dim3 g((unsigned)cdiv(N_NODES, 128), (HEAD * HID) / 16);
        gemm_wmma_async<<<g, 256, 0, stream>>>(xh, W1t, nullptr, nullptr, feat1h,
                                               N_NODES, IN_DIM, HEAD * HID);
    }
    attn_coef<<<N_NODES, HEAD * HID, 2 * HEAD * HID * sizeof(float), stream>>>(
        feat1h, al1, ar1, el, er, HEAD, HID);

    long long NH = (long long)N_NODES * HEAD;
    long long EH = (long long)N_EDGES * HEAD;
    fill_u32<<<(unsigned)cdiv(NH, 256), 256, 0, stream>>>(emax, ORD_NEG_INF, NH);
    fill_u32<<<(unsigned)cdiv(NH, 256), 256, 0, stream>>>((unsigned*)denom, 0u, NH);
    fill_u32<<<(unsigned)cdiv((long long)N_NODES * 256, 256), 256, 0, stream>>>(
        (unsigned*)agg1, 0u, (long long)N_NODES * 256);

    edge_max<<<(unsigned)cdiv(EH, 256), 256, 0, stream>>>(el, er, src, dst, emax, EH, HEAD);
    decode_max<<<(unsigned)cdiv(NH, 256), 256, 0, stream>>>(emax, NH);
    edge_expsum<<<(unsigned)cdiv(EH, 256), 256, 0, stream>>>(
        el, er, src, dst, (const float*)emax, ee, denom, EH, HEAD);
    scatter_msg<<<N_EDGES, HEAD * HID, 0, stream>>>(feat1h, ee, denom, src, dst,
                                                    agg1, HEAD, HID);

    // BN + ReLU -> h1 (f16, overwrites feat1h which is dead after scatter)
    fill_u32<<<2, 256, 0, stream>>>((unsigned*)gsum, 0u, 512);
    bn_stats<<<200, HEAD * HID, 0, stream>>>(agg1, gsum, gsq, N_NODES, HEAD * HID, 500);
    bn_finalize<<<1, 256, 0, stream>>>(gsum, gsq, mu, rstd, HEAD * HID, invN);
    bn_relu_h<<<(unsigned)cdiv((long long)N_NODES * 256, 256), 256, 0, stream>>>(
        agg1, mu, rstd, g1, be1, feat1h, (long long)N_NODES * 256, HEAD * HID);

    // =========================== Layer 2 (H=1, D=64) ========================
    transpose_f16<<<(unsigned)cdiv(HEAD * HID * HID, 256), 256, 0, stream>>>(
        W2, W2t, HEAD * HID, HID);
    {
        dim3 g((unsigned)cdiv(N_NODES, 128), HID / 16);
        gemm_wmma_async<<<g, 256, 0, stream>>>(feat1h, W2t, nullptr, nullptr, feat2h,
                                               N_NODES, HEAD * HID, HID);
    }
    attn_coef<<<N_NODES, HID, 2 * HID * sizeof(float), stream>>>(
        feat2h, al2, ar2, el, er, 1, HID);

    long long NH2 = N_NODES;
    long long EH2 = N_EDGES;
    fill_u32<<<(unsigned)cdiv(NH2, 256), 256, 0, stream>>>(emax, ORD_NEG_INF, NH2);
    fill_u32<<<(unsigned)cdiv(NH2, 256), 256, 0, stream>>>((unsigned*)denom, 0u, NH2);
    fill_u32<<<(unsigned)cdiv((long long)N_NODES * 64, 256), 256, 0, stream>>>(
        (unsigned*)agg2, 0u, (long long)N_NODES * 64);

    edge_max<<<(unsigned)cdiv(EH2, 256), 256, 0, stream>>>(el, er, src, dst, emax, EH2, 1);
    decode_max<<<(unsigned)cdiv(NH2, 256), 256, 0, stream>>>(emax, NH2);
    edge_expsum<<<(unsigned)cdiv(EH2, 256), 256, 0, stream>>>(
        el, er, src, dst, (const float*)emax, ee, denom, EH2, 1);
    scatter_msg<<<N_EDGES, HID, 0, stream>>>(feat2h, ee, denom, src, dst, agg2, 1, HID);

    // BN + ReLU -> h2 (f32, overwrites ee region which is dead after scatter)
    fill_u32<<<1, 256, 0, stream>>>((unsigned*)gsum, 0u, 512);
    bn_stats<<<200, HID, 0, stream>>>(agg2, gsum, gsq, N_NODES, HID, 500);
    bn_finalize<<<1, 64, 0, stream>>>(gsum, gsq, mu, rstd, HID, invN);
    bn_relu_f<<<(unsigned)cdiv((long long)N_NODES * 64, 256), 256, 0, stream>>>(
        agg2, mu, rstd, g2, be2, h2, (long long)N_NODES * 64, HID);

    // =================== L2 normalize -> feat output (f32 + f16) ============
    l2norm64<<<(unsigned)cdiv(N_NODES, 8), 256, 0, stream>>>(h2, feat_out, feath, N_NODES);

    // =================== Output head: out = feat @ Wo + bo ==================
    transpose_f16<<<(unsigned)cdiv(HID * NUM_CLS, 256), 256, 0, stream>>>(
        Wo, Wot, HID, NUM_CLS);
    {
        dim3 g((unsigned)cdiv(N_NODES, 128), (unsigned)cdiv(NUM_CLS, 16));
        gemm_wmma_async<<<g, 256, 0, stream>>>(feath, Wot, bo, out, nullptr,
                                               N_NODES, HID, NUM_CLS);
    }
}
